// CRPS_loss_6305011991216
// MI455X (gfx1250) — compile-verified, matrix operand order checked by reading, not run on hardware
//
#include <hip/hip_runtime.h>
#include <stdint.h>

#ifndef __has_builtin
#define __has_builtin(x) 0
#endif

#if __has_builtin(__builtin_amdgcn_tensor_load_to_lds)
#define USE_TDM 1
#else
#define USE_TDM 0
#endif

#if __has_builtin(__builtin_amdgcn_s_wait_tensorcnt)
#define HAVE_WAIT_TENSOR 1
#else
#define HAVE_WAIT_TENSOR 0
#endif

#if __has_builtin(__builtin_amdgcn_wmma_f32_16x16x4_f32)
#define USE_WMMA_F32X4 1
#else
#define USE_WMMA_F32X4 0
#endif

#if USE_TDM
#if HAVE_WAIT_TENSOR
#define WAIT_TENSOR(n) __builtin_amdgcn_s_wait_tensorcnt(n)
#else
#define WAIT_TENSOR(n) asm volatile("s_wait_tensorcnt " #n ::: "memory")
#endif
#else
#define WAIT_TENSOR(n) do {} while (0)
#endif

typedef __attribute__((ext_vector_type(4))) unsigned int u32x4;
typedef __attribute__((ext_vector_type(8))) int i32x8;
typedef __attribute__((ext_vector_type(4))) int i32x4;
typedef __attribute__((ext_vector_type(2))) float v2f;
typedef __attribute__((ext_vector_type(8))) float v8f;
typedef __attribute__((ext_vector_type(16))) _Float16 v16h;

#define WAVES 8
#define SPB 16                      // samples per block (2 per wave)
#define CHUNK_T 32                  // timesteps per staged chunk
#define DQ 11                       // quantiles per row
#define TMAX 512
#define TZP (TMAX + 4)              // trapz LDS pitch (conflict-free WMMA fetch)
#define CHUNK_ELEMS (CHUNK_T * DQ)  // 352 floats = 1408 bytes (16B aligned)

// CDNA5: low 32 bits of a flat pointer in the LDS aperture are the LDS byte
// address (ISA 10.2 aperture table) — exactly what D#.lds_addr wants.
__device__ __forceinline__ uint32_t lds_byte_offset(const void* p) {
  return (uint32_t)(uintptr_t)p;
}

// Issue one TDM load of a contiguous 1408B chunk (global -> LDS), async.
__device__ __forceinline__ void load_chunk(const float* __restrict__ gsrc,
                                           float* __restrict__ ldst) {
#if USE_TDM
  uint64_t ga = (uint64_t)(uintptr_t)gsrc;
  uint32_t la = lds_byte_offset(ldst);
  u32x4 g0;
  g0[0] = 1u;                                   // count=1, user mode, no gather
  g0[1] = la;                                   // lds_addr (bytes)
  g0[2] = (uint32_t)ga;                         // global_addr[31:0]
  g0[3] = (uint32_t)((ga >> 32) & 0x01FFFFFFu)  // global_addr[56:32]
          | (2u << 30);                         // type = 2 ("image")
  i32x8 g1;
  g1[0] = (int)(2u << 16);                      // data_size = 4B; mask/flags 0
  g1[1] = (int)((CHUNK_ELEMS & 0xFFFF) << 16);  // tensor_dim0[15:0]
  g1[2] = (int)((CHUNK_ELEMS >> 16) | (1 << 16)); // td0 hi | tensor_dim1=1
  g1[3] = (int)((CHUNK_ELEMS & 0xFFFF) << 16);  // td1 hi=0 | tile_dim0=352
  g1[4] = 1;                                    // tile_dim1=1, tile_dim2=0
  g1[5] = CHUNK_ELEMS;                          // tensor_dim0_stride lo32
  g1[6] = (int)((CHUNK_ELEMS & 0xFFFF) << 16);  // s0 hi=0 | td1_stride lo16
  g1[7] = 0;
  i32x4 g2 = {0, 0, 0, 0};                      // group2: dims 2/3 unused (1-D)
  i32x4 g3 = {0, 0, 0, 0};                      // group3: dims 3/4 unused
  i32x8 g4 = {0, 0, 0, 0, 0, 0, 0, 0};          // extra arg of 6-arg clang-23 form
  // clang-23 / therock-10.0 signature: (u32x4, i32x8, i32x4, i32x4, i32x8, cpol)
  __builtin_amdgcn_tensor_load_to_lds(g0, g1, g2, g3, g4, 0);
#else
  const int lane = threadIdx.x & 31;
  for (int i = lane; i < CHUNK_ELEMS; i += 32) ldst[i] = gsrc[i];
#endif
}

__global__ __launch_bounds__(256) void crps_kernel(
    const float* __restrict__ inp, const float* __restrict__ target,
    float* __restrict__ out, int n, int T) {
  __shared__ __align__(16) float stage[WAVES][2][CHUNK_ELEMS];  // 22528 B
  __shared__ float tz[SPB][TZP];                                // 33024 B

  const int tid = threadIdx.x;
  const int wave = tid >> 5;
  const int lane = tid & 31;
  const int blockSample = blockIdx.x * SPB;
  const int waveSample = blockSample + wave * 2;  // first of the wave's 2 rows

  const int cps = T / CHUNK_T;   // chunks per sample (16 for T=512)
  const int nChunks = 2 * cps;   // contiguous stream: 2 consecutive samples
  const float* waveSrc = inp + (size_t)waveSample * T * DQ;

  static const float F[13] = {0.0f, 0.0f, 0.1f, 0.2f, 0.3f, 0.4f, 0.5f,
                              0.6f, 0.7f, 0.8f, 0.9f, 1.0f, 1.0f};

  // Prologue: stage chunk 0 into buffer 0.
  load_chunk(waveSrc, &stage[wave][0][0]);

  for (int c = 0; c < nChunks; ++c) {
    const int buf = c & 1;
    if (c + 1 < nChunks) {
      load_chunk(waveSrc + (size_t)(c + 1) * CHUNK_ELEMS,
                 &stage[wave][buf ^ 1][0]);
      WAIT_TENSOR(1);  // chunk c complete; chunk c+1 may stay in flight
    } else {
      WAIT_TENSOR(0);
    }
    asm volatile("" ::: "memory");  // keep LDS reads below the wait

    const int sLocal = wave * 2 + c / cps;
    const int s = blockSample + sLocal;
    const int t = (c % cps) * CHUNK_T + lane;

    // Row fetch from LDS: stride 11 dwords (odd) -> bank-conflict-free.
    const float* row = &stage[wave][buf][lane * DQ];
    const float obs = target[(size_t)s * T + t];

    float x[13];
#pragma unroll
    for (int j = 0; j < DQ; ++j) x[j + 1] = row[j];
    const bool beyond = x[11] < obs;  // obs past the last forecast quantile
    x[0] = fminf(obs, x[1]);
    x[12] = fmaxf(obs, x[11]);

    float y[13];
#pragma unroll
    for (int j = 0; j < 13; ++j) {
      float o = (x[j] >= obs) ? 1.0f : 0.0f;
      if (j == 12 && beyond) o = 0.0f;  // reference zeroes last column
      const float d = F[j] - o;
      y[j] = d * d;
    }
    float acc = 0.0f;
#pragma unroll
    for (int j = 0; j < 12; ++j) acc += (y[j + 1] + y[j]) * (x[j + 1] - x[j]);
    tz[sLocal][t] = 0.5f * acc;
  }
  __syncthreads();

  // Mean over T as a WMMA GEMV: D = A(16xK tiles of trapz) x ones + C.
  // With B = all-ones every B layout is correct; each D row holds its row sum.
  if (wave == 0) {
    const int half = lane >> 4;  // A f32 16x4 layout: K pairs {0,1} / {2,3}
    const int m = lane & 15;
    const float inv = 1.0f / (float)T;
#if USE_WMMA_F32X4
    v8f acc8 = {0.f, 0.f, 0.f, 0.f, 0.f, 0.f, 0.f, 0.f};
    v2f bones;
    bones[0] = 1.0f;
    bones[1] = 1.0f;
    for (int k0 = 0; k0 < T; k0 += 4) {
      v2f a;
      a[0] = tz[m][k0 + 2 * half];      // lanes 0-15: K=k0;   16-31: K=k0+2
      a[1] = tz[m][k0 + 2 * half + 1];  // lanes 0-15: K=k0+1; 16-31: K=k0+3
      acc8 = __builtin_amdgcn_wmma_f32_16x16x4_f32(
          false, a, false, bones, (short)0, acc8, false, false);
    }
#else
    // Fallback: codegen-confirmed f16 K=32 WMMA (trapz <= ~100, f16-safe-ish).
    v8f acc8 = {0.f, 0.f, 0.f, 0.f, 0.f, 0.f, 0.f, 0.f};
    v16h bones;
#pragma unroll
    for (int i = 0; i < 16; ++i) bones[i] = (_Float16)1.0f;
    for (int k0 = 0; k0 < T; k0 += 32) {
      v16h a;
#pragma unroll
      for (int v = 0; v < 8; ++v) {
        const int kb = (v < 4 ? 2 * v : 16 + 2 * (v - 4)) + (half ? 8 : 0);
        a[2 * v] = (_Float16)tz[m][k0 + kb];
        a[2 * v + 1] = (_Float16)tz[m][k0 + kb + 1];
      }
      acc8 = __builtin_amdgcn_wmma_f32_16x16x32_f16(
          false, a, false, bones, (short)0, acc8, false, false);
    }
#endif
    // C/D layout: VGPR r, lanes 0-15 -> M=r (col=lane); lanes 16-31 -> M=r+8.
    if (lane == 0) {
#pragma unroll
      for (int r = 0; r < 8; ++r) out[blockSample + r] = acc8[r] * inv;
    } else if (lane == 16) {
#pragma unroll
      for (int r = 0; r < 8; ++r) out[blockSample + 8 + r] = acc8[r] * inv;
    }
  }
}

extern "C" void kernel_launch(void* const* d_in, const int* in_sizes, int n_in,
                              void* d_out, int out_size, void* d_ws,
                              size_t ws_size, hipStream_t stream) {
  const float* inp = (const float*)d_in[0];     // (n, T, 11) f32
  const float* target = (const float*)d_in[1];  // (n, T) f32
  float* out = (float*)d_out;                   // (n,) f32
  const int n = out_size;                       // 4096
  const int T = in_sizes[1] / n;                // 512
  const int blocks = n / SPB;                   // 256 blocks x 8 waves
  crps_kernel<<<blocks, 256, 0, stream>>>(inp, target, out, n, T);
}
